// GNNLayer_63178968924284
// MI455X (gfx1250) — matrix operand order repaired and checked by reading, not compiled, not run on hardware
//
#include <hip/hip_runtime.h>

typedef __attribute__((ext_vector_type(16))) _Float16 v16h;
typedef __attribute__((ext_vector_type(8)))  _Float16 v8h;
typedef __attribute__((ext_vector_type(8)))  float    v8f;
typedef __attribute__((ext_vector_type(4)))  float    v4f;

#define IN_F  512
#define OUT_F 256
#define LDS_STRIDE (IN_F + 8)   // 520 halves: stride % 64 banks == 4 -> conflict-free b128 reads
#define LDS_HALVES (OUT_F * LDS_STRIDE)   // 133120 halves = 260 KB (<= 320 KB WGP LDS)

// ---------------------------------------------------------------------------
// WcT[n][k] = (f16) W[k][n]  -- 256 KB, L2-resident, staged into LDS by GEMM
// ---------------------------------------------------------------------------
__global__ void wt_convert_kernel(const float* __restrict__ W,
                                  _Float16* __restrict__ WcT) {
  int tid = blockIdx.x * blockDim.x + threadIdx.x;   // IN_F*OUT_F threads
  int k = tid & (IN_F - 1);
  int n = tid >> 9;                                  // /IN_F
  if (n < OUT_F) WcT[(size_t)n * IN_F + k] = (_Float16)W[(size_t)k * OUT_F + n];
}

// ---------------------------------------------------------------------------
// GEMM: sup(f16)[row][n] = features[row][:] @ W[:][n]
// Block: 8 waves x 16 rows = 128 rows, full 256 out-features per wave
// (16 accumulators) -> features streamed from HBM exactly once.
// W^T staged once per block into padded LDS; 16 per-tile LDS base pointers +
// fully-unrolled k-loop => every ds_load_b128 uses a 16-bit immediate offset
// (zero address VALU on the load path). Feature (B) fragments double-buffered.
// ---------------------------------------------------------------------------
__global__ void __launch_bounds__(256)
gemm_wmma_kernel(const float* __restrict__ F, const _Float16* __restrict__ WcT,
                 _Float16* __restrict__ sup, int nrows) {
  extern __shared__ _Float16 smem[];   // [OUT_F][LDS_STRIDE]

  const int tid  = threadIdx.x;
  const int lane = tid & 31;
  const int wave = tid >> 5;
  const int sel  = lane >> 4;         // half-wave selector
  const int l16  = lane & 15;

  // ---- stage WcT -> LDS (coalesced: consecutive lanes, consecutive 16B) ----
  for (int idx = tid; idx < OUT_F * (IN_F / 8); idx += 256) {
    int n = idx >> 6;                  // row (out-feature)
    int c = (idx & 63) * 8;            // k-chunk
    *(v8h*)(smem + (size_t)n * LDS_STRIDE + c) =
        *(const v8h*)(WcT + (size_t)n * IN_F + c);
  }
  __syncthreads();

  const int row  = blockIdx.x * 128 + wave * 16 + l16;   // output row (B col)
  const int rowc = (row < nrows) ? row : (nrows - 1);    // clamp for loads

  const float* fbase = F + (size_t)rowc * IN_F + sel * 8;

  // 16 per-tile LDS base pointers; all k offsets become DS immediates
  const _Float16* wtile[16];
#pragma unroll
  for (int j = 0; j < 16; ++j)
    wtile[j] = smem + (size_t)(j * 16 + l16) * LDS_STRIDE + sel * 8;

  v8f acc[16];
#pragma unroll
  for (int j = 0; j < 16; ++j)
#pragma unroll
    for (int i = 0; i < 8; ++i) acc[j][i] = 0.0f;

  // prime B double-buffer (k0 = 0)
  v4f f0 = *(const v4f*)(fbase);
  v4f f1 = *(const v4f*)(fbase + 4);
  v4f f2 = *(const v4f*)(fbase + 16);
  v4f f3 = *(const v4f*)(fbase + 20);

#pragma unroll
  for (int k0 = 0; k0 < IN_F; k0 += 32) {
    // convert current B fragment (waits on loads issued a full k-step ago)
    v16h b;
#pragma unroll
    for (int i = 0; i < 4; ++i) {
      b[i]      = (_Float16)f0[i];
      b[4 + i]  = (_Float16)f1[i];
      b[8 + i]  = (_Float16)f2[i];
      b[12 + i] = (_Float16)f3[i];
    }
    // issue next k-step's B loads now; latency hidden behind 16 WMMAs
    if (k0 + 32 < IN_F) {
      f0 = *(const v4f*)(fbase + k0 + 32);
      f1 = *(const v4f*)(fbase + k0 + 36);
      f2 = *(const v4f*)(fbase + k0 + 48);
      f3 = *(const v4f*)(fbase + k0 + 52);
    }
#pragma unroll
    for (int g = 0; g < 4; ++g) {
      // 4 A-fragments from LDS (8 x ds_load_b128, immediate offsets),
      // then 4 back-to-back WMMAs
      v8h lo[4], hi[4];
#pragma unroll
      for (int t = 0; t < 4; ++t) {
        const _Float16* wp = wtile[g * 4 + t] + k0;
        lo[t] = *(const v8h*)(wp);        // K = k0 + sel*8 .. +7
        hi[t] = *(const v8h*)(wp + 16);   // K = k0 + 16 + sel*8 .. +7
      }
#pragma unroll
      for (int t = 0; t < 4; ++t) {
        v16h a;
#pragma unroll
        for (int i = 0; i < 8; ++i) { a[i] = lo[t][i]; a[8 + i] = hi[t][i]; }
        acc[g * 4 + t] = __builtin_amdgcn_wmma_f32_16x16x32_f16(
            false, a, false, b, (short)0, acc[g * 4 + t], false, false);
      }
    }
  }

  // D lanes hold 8 consecutive output features of one row -> one b128 store
  if (row < nrows) {
#pragma unroll
    for (int j = 0; j < 16; ++j) {
      v8h o;
#pragma unroll
      for (int r = 0; r < 8; ++r) o[r] = (_Float16)acc[j][r];
      *(v8h*)(sup + (size_t)row * OUT_F + j * 16 + sel * 8) = o;
    }
  }
}

// ---------------------------------------------------------------------------
// CSR construction: zero counts -> histogram -> scan -> scatter
// ---------------------------------------------------------------------------
__global__ void zero_i32_kernel(int* __restrict__ p, int n) {
  int i = blockIdx.x * blockDim.x + threadIdx.x;
  if (i < n) p[i] = 0;
}

__global__ void count_rows_kernel(const int* __restrict__ er,
                                  int* __restrict__ counts, int E) {
  int e = blockIdx.x * blockDim.x + threadIdx.x;
  if (e < E) atomicAdd(&counts[er[e]], 1);
}

__global__ void __launch_bounds__(1024)
scan_counts_kernel(const int* __restrict__ counts, int* __restrict__ starts,
                   int* __restrict__ cursors, int n) {
  __shared__ int sm[1024];
  __shared__ int running;
  if (threadIdx.x == 0) running = 0;
  __syncthreads();
  for (int base = 0; base < n; base += 1024) {
    int i = base + (int)threadIdx.x;
    int v = (i < n) ? counts[i] : 0;
    sm[threadIdx.x] = v;
    __syncthreads();
    for (int off = 1; off < 1024; off <<= 1) {
      int t = (threadIdx.x >= (unsigned)off) ? sm[threadIdx.x - off] : 0;
      __syncthreads();
      sm[threadIdx.x] += t;
      __syncthreads();
    }
    int excl = sm[threadIdx.x] - v;
    int base_run = running;
    if (i < n) { int s = base_run + excl; starts[i] = s; cursors[i] = s; }
    __syncthreads();
    if (threadIdx.x == 0) running += sm[1023];
    __syncthreads();
  }
  if (threadIdx.x == 0) starts[n] = running;
}

__global__ void scatter_edges_kernel(const int* __restrict__ er,
                                     int* __restrict__ cursors,
                                     int* __restrict__ eidx, int E) {
  int e = blockIdx.x * blockDim.x + threadIdx.x;
  if (e < E) {
    int p = atomicAdd(&cursors[er[e]], 1);
    eidx[p] = e;
  }
}

// ---------------------------------------------------------------------------
// CSR SpMM + ReLU: one wave per row, register accumulation, no fp atomics.
// sup is f16 (51 MB) -> L2-resident gathers; 2 independent gather chains in
// flight per wave to hide L2 latency.
// ---------------------------------------------------------------------------
__global__ void __launch_bounds__(256)
spmm_csr_kernel(const _Float16* __restrict__ sup, const int* __restrict__ starts,
                const int* __restrict__ eidx, const int* __restrict__ ecol,
                const float* __restrict__ eval, float* __restrict__ out,
                int nrows) {
  const int lane = threadIdx.x & 31;
  const int row  = blockIdx.x * 8 + (threadIdx.x >> 5);
  if (row >= nrows) return;

  float acc[8];
#pragma unroll
  for (int j = 0; j < 8; ++j) acc[j] = 0.0f;

  const int s = starts[row];
  const int t = starts[row + 1];

  int p = s;
  for (; p + 1 < t; p += 2) {
    int e0 = eidx[p];
    int e1 = eidx[p + 1];
    int c0 = ecol[e0];
    int c1 = ecol[e1];
    float v0 = eval[e0];
    float v1 = eval[e1];
    v8h g0 = *(const v8h*)(sup + (size_t)c0 * OUT_F + lane * 8);
    v8h g1 = *(const v8h*)(sup + (size_t)c1 * OUT_F + lane * 8);
#pragma unroll
    for (int j = 0; j < 8; ++j) acc[j] = fmaf(v0, (float)g0[j], acc[j]);
#pragma unroll
    for (int j = 0; j < 8; ++j) acc[j] = fmaf(v1, (float)g1[j], acc[j]);
  }
  if (p < t) {
    int e   = eidx[p];
    int c   = ecol[e];
    float v = eval[e];
    v8h g = *(const v8h*)(sup + (size_t)c * OUT_F + lane * 8);
#pragma unroll
    for (int j = 0; j < 8; ++j) acc[j] = fmaf(v, (float)g[j], acc[j]);
  }

  float* orow = out + (size_t)row * OUT_F + lane * 8;
  v4f o0, o1;
#pragma unroll
  for (int j = 0; j < 4; ++j) {
    o0[j] = fmaxf(acc[j], 0.0f);
    o1[j] = fmaxf(acc[4 + j], 0.0f);
  }
  *(v4f*)orow       = o0;
  *((v4f*)orow + 1) = o1;
}

// ---------------------------------------------------------------------------
extern "C" void kernel_launch(void* const* d_in, const int* in_sizes, int n_in,
                              void* d_out, int out_size, void* d_ws, size_t ws_size,
                              hipStream_t stream) {
  const float* features = (const float*)d_in[0];
  const float* weight   = (const float*)d_in[1];
  const int*   edge_row = (const int*)d_in[2];
  const int*   edge_col = (const int*)d_in[3];
  const float* edge_val = (const float*)d_in[4];
  float*       out      = (float*)d_out;

  const int N = in_sizes[0] / IN_F;   // 100000
  const int E = in_sizes[2];          // 3200000

  // Workspace carve-up (256B aligned): sup f16 | WcT f16 | counts | cursors | starts | eidx
  char* ws = (char*)d_ws;
  size_t off = 0;
  auto align256 = [](size_t x) { return (x + 255) & ~(size_t)255; };

  _Float16* sup = (_Float16*)(ws + off); off = align256(off + (size_t)N * OUT_F * sizeof(_Float16));
  _Float16* WcT = (_Float16*)(ws + off); off = align256(off + (size_t)IN_F * OUT_F * sizeof(_Float16));
  int* counts   = (int*)(ws + off);      off = align256(off + (size_t)N * sizeof(int));
  int* cursors  = (int*)(ws + off);      off = align256(off + (size_t)N * sizeof(int));
  int* starts   = (int*)(ws + off);      off = align256(off + (size_t)(N + 1) * sizeof(int));
  int* eidx     = (int*)(ws + off);      off = align256(off + (size_t)E * sizeof(int));

  // 1) W -> WcT (f16, transposed)
  wt_convert_kernel<<<(IN_F * OUT_F + 255) / 256, 256, 0, stream>>>(weight, WcT);

  // 2) dense GEMM via v_wmma_f32_16x16x32_f16; W^T in 260 KB LDS per block
  size_t lds_bytes = (size_t)LDS_HALVES * sizeof(_Float16);
  gemm_wmma_kernel<<<(N + 127) / 128, 256, lds_bytes, stream>>>(features, WcT, sup, N);

  // 3) CSR build
  zero_i32_kernel<<<(N + 255) / 256, 256, 0, stream>>>(counts, N);
  count_rows_kernel<<<(E + 255) / 256, 256, 0, stream>>>(edge_row, counts, E);
  scan_counts_kernel<<<1, 1024, 0, stream>>>(counts, starts, cursors, N);
  scatter_edges_kernel<<<(E + 255) / 256, 256, 0, stream>>>(edge_row, cursors, eidx, E);

  // 4) SpMM + ReLU, one wave per row, no fp atomics
  spmm_csr_kernel<<<(N + 7) / 8, 256, 0, stream>>>(sup, starts, eidx, edge_col,
                                                   edge_val, out, N);
}
